// NPA_27006754357605
// MI455X (gfx1250) — matrix operand order repaired and checked by reading, not compiled, not run on hardware
//
#include <hip/hip_runtime.h>
#include <hip/hip_bf16.h>
#include <math.h>

typedef __attribute__((ext_vector_type(16))) _Float16 v16h;
typedef __attribute__((ext_vector_type(8)))  float    v8f;

constexpr int kS = 512;   // state count
constexpr int kA = 32;    // actions
constexpr int kB = 128;   // batch
constexpr int kT = 64;    // trajectory length

// ---------------------------------------------------------------------------
// Kernel 1: row softmax of (theta+mask), writing TRANSPOSED Pt[a][t][s] so the
// 64-step hot loop needs no transpose at all: Pt rows are the WMMA A operand.
// One-time scattered 2B stores; each action block is a contiguous 512 KB.
// ---------------------------------------------------------------------------
__global__ __launch_bounds__(128) void softmax_kernel(
    const float* __restrict__ theta, const float* __restrict__ mask,
    _Float16* __restrict__ Pt)
{
    int r = blockIdx.x;            // r = s*kA + a  (input row index)
    int s = r / kA;
    int a = r % kA;
    const float* tin  = theta + (size_t)r * kS;
    const float* min_ = mask  + (size_t)r * kS;
    _Float16* pout = Pt + (size_t)a * kS * kS + s;   // column s of action a

    __shared__ float red[128];
    int tid = threadIdx.x;

    float vals[4];
    float mx = -1e30f;
#pragma unroll
    for (int i = 0; i < 4; ++i) {
        int j = tid + i * 128;
        float v = tin[j] + min_[j];
        vals[i] = v;
        mx = fmaxf(mx, v);
    }
    red[tid] = mx; __syncthreads();
    for (int off = 64; off > 0; off >>= 1) {
        if (tid < off) red[tid] = fmaxf(red[tid], red[tid + off]);
        __syncthreads();
    }
    mx = red[0];
    __syncthreads();

    float sum = 0.f;
#pragma unroll
    for (int i = 0; i < 4; ++i) {
        vals[i] = __expf(vals[i] - mx);
        sum += vals[i];
    }
    red[tid] = sum; __syncthreads();
    for (int off = 64; off > 0; off >>= 1) {
        if (tid < off) red[tid] += red[tid + off];
        __syncthreads();
    }
    float inv = 1.0f / red[0];
#pragma unroll
    for (int i = 0; i < 4; ++i)
        pout[(size_t)(tid + i * 128) * kS] = (_Float16)(vals[i] * inv);
}

// ---------------------------------------------------------------------------
// Kernel 2: per-step counting sort of batches by action.
// ---------------------------------------------------------------------------
__global__ __launch_bounds__(128) void group_kernel(
    const int* __restrict__ acts, int* __restrict__ order, int* __restrict__ offsets)
{
    int t = blockIdx.x;
    int b = threadIdx.x;                 // 128 threads = batches
    __shared__ int cnt[kA];
    __shared__ int offs[kA + 1];
    if (b < kA) cnt[b] = 0;
    __syncthreads();
    int a = acts[b * kT + t];
    int rank = atomicAdd(&cnt[a], 1);
    __syncthreads();
    if (b == 0) {
        int acc = 0;
        for (int i = 0; i < kA; ++i) { offs[i] = acc; acc += cnt[i]; }
        offs[kA] = acc;
    }
    __syncthreads();
    order[t * kB + offs[a] + rank] = b;
    if (b <= kA) offsets[t * (kA + 1) + b] = offs[b];
}

// ---------------------------------------------------------------------------
// Kernel 3: convert init_dist (f32, already normalized) -> f16, zero logacc.
// ---------------------------------------------------------------------------
__global__ __launch_bounds__(256) void init_kernel(
    const float* __restrict__ init_dist, _Float16* __restrict__ dist,
    float* __restrict__ logacc)
{
    int i = blockIdx.x * blockDim.x + threadIdx.x;   // kB*kS threads
    dist[i] = (_Float16)init_dist[i];
    if (i < kB) logacc[i] = 0.f;
}

// ---------------------------------------------------------------------------
// Kernel 4 (per step): grouped GEMM via WMMA, transposed formulation.
//   C'[t_out, b] = sum_s Pt[a][t_out][s] * dist[b][s]
// A = Pt rows (M = t_out), B = dist rows as columns (N = group batches).
// Both fragments load as contiguous 16B chunks from global -> no LDS, no
// barriers, no gathers. Each wave does TWO M-tiles sharing one B fragment.
// Block = 1 wave. Grid = (kS/32 M-pairs, kA actions).
// ---------------------------------------------------------------------------
__global__ __launch_bounds__(32) void step_gemm_kernel(
    const _Float16* __restrict__ Pt, const _Float16* __restrict__ dist,
    const int* __restrict__ order, const int* __restrict__ offsets,
    float* __restrict__ raw, int t)
{
    int a  = blockIdx.y;
    int g0 = offsets[t * (kA + 1) + a];
    int g1 = offsets[t * (kA + 1) + a + 1];
    int cnt = g1 - g0;
    if (cnt == 0) return;                       // wave-uniform exit

    const int* ord = order + t * kB + g0;
    int lane = threadIdx.x;
    int lo   = lane & 15;
    int hi   = lane >> 4;

    int mbase0 = blockIdx.x * 32;               // this wave's two t_out tiles
    const _Float16* Arow0 = Pt + (size_t)a * kS * kS + (size_t)(mbase0 + lo) * kS;
    const _Float16* Arow1 = Arow0 + (size_t)16 * kS;

    // 16-bit A 16x32 layout: lanes 0-15 elems = K{0-7,16-23}; lanes 16-31 = K{8-15,24-31}
    int ka0 = hi ? 8 : 0;
    int ka1 = hi ? 24 : 16;
    // 16-bit B 32x16 layout: lane holds fixed N; elems = K{0-7,16-23} (lo) / K{8-15,24-31} (hi), same as A
    int kb  = hi ? 8 : 0;

    for (int n0 = 0; n0 < cnt; n0 += 16) {
        int nb = n0 + lo;
        int bn = ord[nb < cnt ? nb : cnt - 1];  // clamped; padded cols never stored
        const _Float16* Brow = dist + (size_t)bn * kS + kb;

        v8f c0 = {};
        v8f c1 = {};
        for (int kt = 0; kt < 16; ++kt) {
            int kbase = kt * 32;
            union { v16h v; uint4 q[2]; } bf, a0, a1;
            bf.q[0] = *(const uint4*)(Brow + kbase);
            bf.q[1] = *(const uint4*)(Brow + kbase + 16);
            a0.q[0] = *(const uint4*)(Arow0 + kbase + ka0);
            a0.q[1] = *(const uint4*)(Arow0 + kbase + ka1);
            a1.q[0] = *(const uint4*)(Arow1 + kbase + ka0);
            a1.q[1] = *(const uint4*)(Arow1 + kbase + ka1);

            c0 = __builtin_amdgcn_wmma_f32_16x16x32_f16(
                     false, a0.v, false, bf.v, (short)0, c0, false, false);
            c1 = __builtin_amdgcn_wmma_f32_16x16x32_f16(
                     false, a1.v, false, bf.v, (short)0, c1, false, false);
        }

        // C layout: lane holds N=lo, elems i -> M = (hi?8:0)+i  (consecutive t_out)
        if (nb < cnt) {
            float* r0 = raw + (size_t)bn * kS + mbase0 + (hi ? 8 : 0);
            float* r1 = r0 + 16;
#pragma unroll
            for (int i = 0; i < 8; ++i) { r0[i] = c0[i]; r1[i] = c1[i]; }
        }
    }
}

// ---------------------------------------------------------------------------
// Kernel 5 (per step): renormalize dist, accumulate log(scale) in f32.
// Keeps f16 WMMA inputs well-scaled across 64 chained products.
// ---------------------------------------------------------------------------
__global__ __launch_bounds__(256) void normalize_kernel(
    const float* __restrict__ raw, _Float16* __restrict__ dist,
    float* __restrict__ logacc)
{
    int b = blockIdx.x;
    int tid = threadIdx.x;                  // 256 threads, 2 elems each
    __shared__ float red[256];
    const float* r = raw + (size_t)b * kS;
    float v0 = r[tid], v1 = r[tid + 256];
    red[tid] = v0 + v1; __syncthreads();
    for (int off = 128; off > 0; off >>= 1) {
        if (tid < off) red[tid] += red[tid + off];
        __syncthreads();
    }
    float tot = fmaxf(red[0], 1e-37f);
    float inv = 1.0f / tot;
    if (tid == 0) logacc[b] += __logf(tot);
    dist[(size_t)b * kS + tid]       = (_Float16)(v0 * inv);
    dist[(size_t)b * kS + tid + 256] = (_Float16)(v1 * inv);
}

// ---------------------------------------------------------------------------
// Kernel 6: loss = mean_b( -log(exp(logacc_b) + 1e-8) )
// ---------------------------------------------------------------------------
__global__ __launch_bounds__(128) void loss_kernel(
    const float* __restrict__ logacc, float* __restrict__ out)
{
    __shared__ float red[128];
    int b = threadIdx.x;
    float p = __expf(logacc[b]);
    red[b] = -__logf(p + 1e-8f);
    __syncthreads();
    for (int off = 64; off > 0; off >>= 1) {
        if (b < off) red[b] += red[b + off];
        __syncthreads();
    }
    if (b == 0) out[0] = red[0] / (float)kB;
}

// ---------------------------------------------------------------------------
extern "C" void kernel_launch(void* const* d_in, const int* in_sizes, int n_in,
                              void* d_out, int out_size, void* d_ws, size_t ws_size,
                              hipStream_t stream)
{
    const float* theta     = (const float*)d_in[0];   // (S, A, S) f32
    const float* mask      = (const float*)d_in[1];   // (S, A, S) f32
    const float* init_dist = (const float*)d_in[2];   // (B, S)   f32
    const int*   acts      = (const int*)d_in[3];     // (B, T)   int32
    float* out = (float*)d_out;

    char* ws = (char*)d_ws;
    size_t off = 0;
    _Float16* Pt   = (_Float16*)(ws + off); off += (size_t)kA * kS * kS * sizeof(_Float16); // 16.8 MB
    _Float16* dist = (_Float16*)(ws + off); off += (size_t)kB * kS * sizeof(_Float16);      // 128 KB
    float* raw     = (float*)(ws + off);    off += (size_t)kB * kS * sizeof(float);         // 256 KB
    int* order     = (int*)(ws + off);      off += (size_t)kT * kB * sizeof(int);           // 32 KB
    int* offsets   = (int*)(ws + off);      off += (size_t)kT * (kA + 1) * sizeof(int);     // 8.4 KB
    float* logacc  = (float*)(ws + off);    off += (size_t)kB * sizeof(float);

    softmax_kernel<<<kS * kA, 128, 0, stream>>>(theta, mask, Pt);
    group_kernel<<<kT, kB, 0, stream>>>(acts, order, offsets);
    init_kernel<<<(kB * kS) / 256, 256, 0, stream>>>(init_dist, dist, logacc);

    for (int t = 0; t < kT; ++t) {
        dim3 grid(kS / 32, kA);   // 16 M-pair tiles x 32 actions, 1 wave per block
        step_gemm_kernel<<<grid, 32, 0, stream>>>(Pt, dist, order, offsets, raw, t);
        normalize_kernel<<<kB, 256, 0, stream>>>(raw, dist, logacc);
    }
    loss_kernel<<<1, kB, 0, stream>>>(logacc, out);
}
